// LSSModel_40020505264286
// MI455X (gfx1250) — matrix operand order repaired and checked by reading, not compiled, not run on hardware
//
#include <hip/hip_runtime.h>
#include <hip/hip_bf16.h>

// ---------------- problem constants ----------------
#define BB 2
#define NN 6
#define CC 64
#define HH 32
#define WW 88
#define DD 48
#define BEVW 200
#define BEVH 200
#define NPIX (BB * BEVH * BEVW)        // 80000 output pixels
#define C1OUT 128
#define NCLS 10
#define KTOT (9 * CC)                   // 576
#define KCHUNKS (KTOT / 32)             // 18
#define K2CHUNKS (C1OUT / 32)           // 4
#define NTILES (C1OUT / 16)             // 8
#define MTILES (NPIX / 16)              // 5000
#define PADW 202

typedef float v8f  __attribute__((ext_vector_type(8)));
typedef __bf16 v16bf __attribute__((ext_vector_type(16)));

__device__ __forceinline__ unsigned short f32_to_bf16(float f) {
    unsigned int u = __float_as_uint(f);
    unsigned int lsb = (u >> 16) & 1u;
    u += 0x7fffu + lsb;                 // round to nearest even
    return (unsigned short)(u >> 16);
}

// ---------------- 0: zero the BEV accumulator ----------------
__global__ void zero_bev_k(float* __restrict__ bev, int n) {
    int i = blockIdx.x * blockDim.x + threadIdx.x;
    if (i < n) bev[i] = 0.0f;
}

// ---------------- 1: depth softmax + lift + splat ----------------
// one 64-thread block per camera pixel; bev is [B][200][200][64] (NHWC)
__global__ void lift_splat_k(const float* __restrict__ feat,
                             const float* __restrict__ intr,
                             const float* __restrict__ extr,
                             const float* __restrict__ depth_w,
                             const float* __restrict__ depth_b,
                             float* __restrict__ bev) {
    int p = blockIdx.x;                 // 0 .. B*N*H*W-1
    int t = threadIdx.x;                // 0 .. 63
    int w = p % WW;  int tmp = p / WW;
    int h = tmp % HH; tmp /= HH;
    int n = tmp % NN; int b = tmp / NN;

    __shared__ float f[CC];
    __shared__ float logits[DD];
    __shared__ float prob[DD];
    __shared__ int   cell[DD];
    __shared__ int   mcell[DD];
    __shared__ float mprob[DD];
    __shared__ float inv_sum;

    // CDNA5 async DMA: each lane pulls its channel straight into LDS,
    // bypassing VGPRs; fenced with ASYNCcnt then the workgroup barrier.
    {
        const float* gp = feat + ((((size_t)b * NN + n) * CC + t) * HH + h) * WW + w;
        unsigned lds_addr = (unsigned)(size_t)(void*)&f[t];
        asm volatile("global_load_async_to_lds_b32 %0, %1, off"
                     :
                     : "v"(lds_addr), "v"(gp)
                     : "memory");
        asm volatile("s_wait_asynccnt 0x0" ::: "memory");
    }
    __syncthreads();

    if (t < DD) {
        const float* wr = depth_w + t * CC;
        float acc = depth_b[t];
        #pragma unroll 8
        for (int c = 0; c < CC; ++c) acc += f[c] * wr[c];
        logits[t] = acc;
    }
    __syncthreads();
    if (t == 0) {
        float m = logits[0];
        for (int d = 1; d < DD; ++d) m = fmaxf(m, logits[d]);
        float s = 0.0f;
        for (int d = 0; d < DD; ++d) { float e = expf(logits[d] - m); prob[d] = e; s += e; }
        inv_sum = 1.0f / s;
    }
    __syncthreads();
    if (t < DD) {
        prob[t] *= inv_sum;
        // analytic K^-1 * [x,y,1]
        const float* K = intr + (b * NN + n) * 9;
        float fx = K[0], cx = K[2], fy = K[4], cy = K[5];
        float rx = ((float)w - cx) / fx;
        float ry = ((float)h - cy) / fy;
        float depth = 1.0f + (float)t;          // linspace(1,48,48)
        float X = rx * depth, Y = ry * depth, Z = depth;
        const float* E = extr + (b * NN + n) * 16;
        float wx = E[0] * X + E[1] * Y + E[2]  * Z + E[3];
        float wy = E[4] * X + E[5] * Y + E[6]  * Z + E[7];
        // homogeneous w == 1 exactly (row 3 is [0,0,0,1])
        int xi = (int)((wx - (-50.0f)) / 100.0f * (float)BEVW);  // trunc toward 0
        int yi = (int)((wy - (-50.0f)) / 100.0f * (float)BEVH);
        bool valid = (xi >= 0) & (xi < BEVW) & (yi >= 0) & (yi < BEVH);
        cell[t] = valid ? (yi * BEVW + xi) : -1;
    }
    __syncthreads();
    // merge runs of consecutive depth bins that land in the same BEV cell:
    // one atomic per run instead of one per bin
    if (t < DD) {
        bool leader = (t == 0) || (cell[t] != cell[t - 1]);
        int  c  = cell[t];
        float ps = 0.0f;
        if (leader && c >= 0) {
            ps = prob[t];
            for (int e = t + 1; e < DD && cell[e] == c; ++e) ps += prob[e];
        }
        mcell[t] = (leader && c >= 0) ? c : -1;
        mprob[t] = ps;
    }
    __syncthreads();

    float fv = f[t];
    for (int d = 0; d < DD; ++d) {
        int cIdx = mcell[d];
        if (cIdx >= 0) {
            atomicAdd(bev + ((size_t)b * (BEVH * BEVW) + cIdx) * CC + t, fv * mprob[d]);
        }
    }
}

// ---------------- 2a: pack conv1 weights into WMMA B-fragment order ---------
// w1pack[ntile][kchunk][lane][e]  (bf16), k = kchunk*32 + (lane>=16?8:0) + (e&7) + (e>=8?16:0)
__global__ void pack_w1_k(const float* __restrict__ c1w, unsigned short* __restrict__ w1pack) {
    int idx = blockIdx.x * blockDim.x + threadIdx.x;
    if (idx >= NTILES * KCHUNKS * 32 * 16) return;
    int e     = idx & 15;
    int lane  = (idx >> 4) & 31;
    int kc    = (idx >> 9) % KCHUNKS;
    int nt    = (idx >> 9) / KCHUNKS;
    int k     = kc * 32 + ((lane >> 4) << 3) + (e & 7) + ((e >> 3) << 4);
    int cout  = nt * 16 + (lane & 15);
    int tap   = k >> 6;                 // k / 64
    int cin   = k & 63;
    int ky    = tap / 3, kx = tap % 3;
    float v = c1w[((cout * CC + cin) * 3 + ky) * 3 + kx];  // OIHW
    w1pack[idx] = f32_to_bf16(v);
}

// ---------------- 2b: pack conv2 weights (N padded 10 -> 16) ----------------
// w2pack[kchunk][lane][e] (bf16), k(=cin) = kchunk*32 + (lane>=16?8:0) + (e&7) + (e>=8?16:0)
__global__ void pack_w2_k(const float* __restrict__ c2w, unsigned short* __restrict__ w2pack) {
    int idx = blockIdx.x * blockDim.x + threadIdx.x;
    if (idx >= K2CHUNKS * 32 * 16) return;
    int e    = idx & 15;
    int lane = (idx >> 4) & 31;
    int kc   = (idx >> 9);
    int k    = kc * 32 + ((lane >> 4) << 3) + (e & 7) + ((e >> 3) << 4);
    int cout = lane & 15;
    float v = (cout < NCLS) ? c2w[cout * C1OUT + k] : 0.0f;
    w2pack[idx] = f32_to_bf16(v);
}

// ---------------- 3: pad + convert BEV f32 -> bf16 [B][202][202][64] --------
__global__ void pad_bev_k(const float* __restrict__ bev, unsigned short* __restrict__ pbev) {
    int idx = blockIdx.x * blockDim.x + threadIdx.x;
    const int total = BB * PADW * PADW * CC;
    if (idx >= total) return;
    int c  = idx & (CC - 1);
    int pp = idx >> 6;
    int px = pp % PADW; pp /= PADW;
    int py = pp % PADW;
    int b  = pp / PADW;
    unsigned short v = 0;
    if (py >= 1 && py <= BEVH && px >= 1 && px <= BEVW) {
        v = f32_to_bf16(bev[(((size_t)b * BEVH + (py - 1)) * BEVW + (px - 1)) * CC + c]);
    }
    pbev[idx] = v;
}

// ---------------- 4: conv1 3x3 + bias + relu as implicit GEMM via WMMA ------
// one wave (32 threads) computes a 16(pixels) x 16(cout) tile; h1 stored bf16
__global__ void conv1_wmma_k(const unsigned short* __restrict__ pbev,
                             const unsigned short* __restrict__ w1pack,
                             const float* __restrict__ c1b,
                             unsigned short* __restrict__ h1) {
    int lane = threadIdx.x;             // 0..31
    int tm = blockIdx.x;                // 0..4999
    int tn = blockIdx.y;                // 0..7
    int row = lane & 15;
    int hi  = lane >> 4;                // selects K sub-range per ISA A layout

    int m = tm * 16 + row;              // output pixel handled by this lane's A row
    int b = m / (BEVH * BEVW);
    int rem = m % (BEVH * BEVW);
    int y = rem / BEVW, x = rem % BEVW;

    v8f acc = {};
    #pragma unroll
    for (int kc = 0; kc < KCHUNKS; ++kc) {
        int k0   = kc * 32;
        int tap  = k0 >> 6;             // chunk lies entirely within one tap
        int ky   = tap / 3, kx = tap % 3;
        int cin0 = k0 & 63;             // 0 or 32

        // A: lanes hold K offsets {hi*8..hi*8+7} and {16+hi*8..16+hi*8+7}
        const unsigned short* ap =
            pbev + (((size_t)b * PADW + (y + ky)) * PADW + (x + kx)) * CC + cin0 + hi * 8;
        union { v16bf v; uint4 q[2]; } A;
        A.q[0] = *(const uint4*)(ap);
        A.q[1] = *(const uint4*)(ap + 16);

        // B: pre-packed fragment, fully contiguous per lane
        const unsigned short* bp = w1pack + (((size_t)(tn * KCHUNKS + kc) * 32 + lane) * 16);
        union { v16bf v; uint4 q[2]; } Bm;
        Bm.q[0] = *(const uint4*)(bp);
        Bm.q[1] = *(const uint4*)(bp + 8);

        acc = __builtin_amdgcn_wmma_f32_16x16x32_bf16(
            false, A.v, false, Bm.v, (short)0, acc, false, false);
    }

    // epilogue: element r of lane -> (M = hi*8 + r, N = lane&15)
    int cout = tn * 16 + row;
    float bias = c1b[cout];
    #pragma unroll
    for (int r = 0; r < 8; ++r) {
        int pixel = tm * 16 + hi * 8 + r;
        float v = acc[r] + bias;
        h1[(size_t)pixel * C1OUT + cout] = f32_to_bf16(v > 0.0f ? v : 0.0f);
    }
}

// ---------------- 5: conv2 1x1 (128 -> 10) via WMMA, NCHW f32 output --------
// one wave per 16-pixel tile; single N tile (classes 0..9 live, 10..15 padded)
__global__ void conv2_wmma_k(const unsigned short* __restrict__ h1,
                             const unsigned short* __restrict__ w2pack,
                             const float* __restrict__ b2,
                             float* __restrict__ out) {
    int lane = threadIdx.x;
    int tm = blockIdx.x;                // 0..4999
    int row = lane & 15;
    int hi  = lane >> 4;

    int m = tm * 16 + row;
    const unsigned short* base = h1 + (size_t)m * C1OUT;

    v8f acc = {};
    #pragma unroll
    for (int kc = 0; kc < K2CHUNKS; ++kc) {
        const unsigned short* ap = base + kc * 32 + hi * 8;
        union { v16bf v; uint4 q[2]; } A;
        A.q[0] = *(const uint4*)(ap);
        A.q[1] = *(const uint4*)(ap + 16);

        const unsigned short* bp = w2pack + ((size_t)(kc * 32 + lane) * 16);
        union { v16bf v; uint4 q[2]; } Bm;
        Bm.q[0] = *(const uint4*)(bp);
        Bm.q[1] = *(const uint4*)(bp + 8);

        acc = __builtin_amdgcn_wmma_f32_16x16x32_bf16(
            false, A.v, false, Bm.v, (short)0, acc, false, false);
    }

    int cls = row;                       // N index of this lane's D column
    if (cls < NCLS) {
        float bias = b2[cls];
        #pragma unroll
        for (int r = 0; r < 8; ++r) {
            int pixel = tm * 16 + hi * 8 + r;
            int b  = pixel / (BEVH * BEVW);
            int yx = pixel % (BEVH * BEVW);
            out[((size_t)b * NCLS + cls) * (BEVH * BEVW) + yx] = acc[r] + bias;
        }
    }
}

// ---------------- launcher ----------------
extern "C" void kernel_launch(void* const* d_in, const int* in_sizes, int n_in,
                              void* d_out, int out_size, void* d_ws, size_t ws_size,
                              hipStream_t stream) {
    const float* feat = (const float*)d_in[0];
    const float* intr = (const float*)d_in[1];
    const float* extr = (const float*)d_in[2];
    const float* dw   = (const float*)d_in[3];
    const float* db   = (const float*)d_in[4];
    const float* c1w  = (const float*)d_in[5];
    const float* c1b  = (const float*)d_in[6];
    const float* c2w  = (const float*)d_in[7];
    const float* c2b  = (const float*)d_in[8];
    float* out = (float*)d_out;

    char* ws = (char*)d_ws;
    size_t off = 0;
    auto take = [&](size_t bytes) -> void* {
        void* p = ws + off;
        off += (bytes + 255) & ~(size_t)255;
        return p;
    };
    float*          bev    = (float*)take((size_t)BB * BEVH * BEVW * CC * sizeof(float));
    unsigned short* pbev   = (unsigned short*)take((size_t)BB * PADW * PADW * CC * sizeof(unsigned short));
    unsigned short* w1pack = (unsigned short*)take((size_t)NTILES * KCHUNKS * 32 * 16 * sizeof(unsigned short));
    unsigned short* w2pack = (unsigned short*)take((size_t)K2CHUNKS * 32 * 16 * sizeof(unsigned short));
    unsigned short* h1     = (unsigned short*)take((size_t)NPIX * C1OUT * sizeof(unsigned short));

    const int bev_n = BB * BEVH * BEVW * CC;                      // 5,120,000
    zero_bev_k<<<(bev_n + 255) / 256, 256, 0, stream>>>(bev, bev_n);

    lift_splat_k<<<BB * NN * HH * WW, 64, 0, stream>>>(feat, intr, extr, dw, db, bev);

    const int pk1_n = NTILES * KCHUNKS * 32 * 16;                 // 73,728
    pack_w1_k<<<(pk1_n + 255) / 256, 256, 0, stream>>>(c1w, w1pack);

    const int pk2_n = K2CHUNKS * 32 * 16;                         // 2,048
    pack_w2_k<<<(pk2_n + 255) / 256, 256, 0, stream>>>(c2w, w2pack);

    const int pad_n = BB * PADW * PADW * CC;                      // 5,222,912
    pad_bev_k<<<(pad_n + 255) / 256, 256, 0, stream>>>(bev, pbev);

    conv1_wmma_k<<<dim3(MTILES, NTILES), 32, 0, stream>>>(pbev, w1pack, c1b, h1);

    conv2_wmma_k<<<MTILES, 32, 0, stream>>>(h1, w2pack, c2b, out);
}
